// HyperSequenceMemoryUpdater_67104569033108
// MI455X (gfx1250) — compile-verified, matrix operand order checked by reading, not run on hardware
//
#include <hip/hip_runtime.h>
#include <hip/hip_bf16.h>
#include <stdint.h>

typedef __attribute__((ext_vector_type(16))) __bf16 v16bf;
typedef __attribute__((ext_vector_type(8)))  float  v8f;

#define N_TOTAL_   1000000
#define N_GATHER_  200000
#define M_UPDATE_  131072
#define MSG_DIM_   512
#define MEM_DIM_   256

#define BM        128          // message rows per block
#define TPB       512          // 16 waves
#define SX_STR    520          // bf16 elems per msg row in LDS (512 + 8 pad)
#define SH_STR    264          // bf16 elems per h row in LDS  (256 + 8 pad)

// LDS byte offsets (dynamic shared)
#define SX_OFF    0
#define SX_BYTES  (BM * SX_STR * 2)          // 133120
#define SH_OFF    SX_BYTES
#define SH_BYTES  (BM * SH_STR * 2)          // 67584
#define RIDX_OFF  (SH_OFF + SH_BYTES)        // 200704
#define OROW_OFF  (RIDX_OFF + BM * 4)        // 201216
#define SMEM_BYTES (OROW_OFF + BM * 4)       // 201728

struct Pack32 { uint4 a; uint4 b; };

__device__ __forceinline__ unsigned short f2bf(float f) {
  unsigned int u = __float_as_uint(f);
  return (unsigned short)((u + 0x7FFFu + ((u >> 16) & 1u)) >> 16);
}

// Load a 16-element bf16 fragment: 8 elems at p[0..7], 8 elems at p[16..23].
// (Per documented 16-bit A layout: lane-half selects the K-chunk via 'koff'
//  folded into p by the caller; B uses the mirrored layout so the same
//  loader works for both operands.)
__device__ __forceinline__ v16bf ldfrag(const unsigned short* p) {
  Pack32 t;
  t.a = *(const uint4*)(p);
  t.b = *(const uint4*)(p + 16);
  return __builtin_bit_cast(v16bf, t);
}

__device__ __forceinline__ float sigmoidf_(float x) {
  return 1.0f / (1.0f + __expf(-x));
}

// ---------------------------------------------------------------- prologue
__global__ void tgn_convert_w_kernel(const float* __restrict__ wih,
                                     const float* __restrict__ whh,
                                     unsigned short* __restrict__ wih_bf,
                                     unsigned short* __restrict__ whh_bf) {
  int i = blockIdx.x * blockDim.x + threadIdx.x;
  const int NI = 768 * 512;
  const int NH = 768 * 256;
  if (i < NI) wih_bf[i] = f2bf(wih[i]);
  if (i < NH) whh_bf[i] = f2bf(whh[i]);
}

// ------------------------------------------------------------------- tail
__global__ void tgn_tail_kernel(const float* __restrict__ memory_table,
                                const float* __restrict__ last_update,
                                const float* __restrict__ timestamps,
                                const int*   __restrict__ node_ids,
                                float* __restrict__ d_out) {
  int i = blockIdx.x * blockDim.x + threadIdx.x;
  const int ROWS  = N_GATHER_ - M_UPDATE_;       // 68928 gather-only rows
  const int ELTS4 = ROWS * (MEM_DIM_ / 4);       // float4 per row = 64
  if (i < ELTS4) {
    int m  = M_UPDATE_ + (i >> 6);
    int c4 = (i & 63) << 2;
    float4 v = *(const float4*)(memory_table + (size_t)node_ids[m] * MEM_DIM_ + c4);
    *(float4*)(d_out + (size_t)m * MEM_DIM_ + c4) = v;
  }
  if (i < N_GATHER_) {
    float lu = (i < M_UPDATE_) ? timestamps[i] : last_update[node_ids[i]];
    d_out[(size_t)N_GATHER_ * MEM_DIM_ + i] = lu;
  }
}

// ------------------------------------------------------------- main GEMM
// Writes PRE-LayerNorm GRU output to d_out (scattered rows); a follow-up
// kernel normalizes those rows in place.
__global__ __launch_bounds__(TPB)
void tgn_gru_wmma_kernel(const float* __restrict__ memory_table,
                         const float* __restrict__ messages,
                         const unsigned short* __restrict__ wih_bf,
                         const unsigned short* __restrict__ whh_bf,
                         const float* __restrict__ b_ih,
                         const float* __restrict__ b_hh,
                         const int*   __restrict__ node_ids,
                         const int*   __restrict__ tu_ids,
                         float* __restrict__ out_mem) {
  extern __shared__ char smem[];
  unsigned short* SX   = (unsigned short*)(smem + SX_OFF);
  unsigned short* SH   = (unsigned short*)(smem + SH_OFF);
  int*            ridx = (int*)(smem + RIDX_OFF);
  int*            orow = (int*)(smem + OROW_OFF);

  const int tid    = threadIdx.x;
  const int m_base = blockIdx.x * BM;

  // --- phase 0: resolve gather rows / output rows for this block
  if (tid < BM) {
    int t = tu_ids[m_base + tid];
    orow[tid] = t;
    ridx[tid] = node_ids[t];
  }
  __syncthreads();

  // --- phase 1: stage messages (f32 -> bf16) into LDS
  for (int i = tid; i < BM * (MSG_DIM_ / 4); i += TPB) {
    int r  = i >> 7;                 // MSG_DIM_/4 == 128
    int c4 = (i & 127) << 2;
    float4 v = *(const float4*)(messages + (size_t)(m_base + r) * MSG_DIM_ + c4);
    unsigned short* d = SX + r * SX_STR + c4;
    d[0] = f2bf(v.x); d[1] = f2bf(v.y); d[2] = f2bf(v.z); d[3] = f2bf(v.w);
  }
  // stage gathered h (f32 -> bf16) into LDS
  for (int i = tid; i < BM * (MEM_DIM_ / 4); i += TPB) {
    int r  = i >> 6;                 // MEM_DIM_/4 == 64
    int c4 = (i & 63) << 2;
    float4 v = *(const float4*)(memory_table + (size_t)ridx[r] * MEM_DIM_ + c4);
    unsigned short* d = SH + r * SH_STR + c4;
    d[0] = f2bf(v.x); d[1] = f2bf(v.y); d[2] = f2bf(v.z); d[3] = f2bf(v.w);
  }
  __syncthreads();

  // --- phase 2: WMMA GEMMs + fused GRU gates
  const int lane = tid & 31;
  const int w    = tid >> 5;
  const int mg   = w & 3;            // M-group: 32 rows (2 M-tiles)
  const int ng   = w >> 2;           // N-group: 64 output columns (4 tiles)
  const int cl   = lane & 15;        // col-in-tile / row-in-tile
  const int hi   = lane >> 4;        // lane half
  const int koff = hi << 3;          // K chunk select: 0 or 8

  #pragma unroll
  for (int j = 0; j < 4; ++j) {
    const int jcol = ng * 64 + j * 16 + cl;     // output column 0..255
    v8f axr[2] = {v8f{}, v8f{}}, axz[2] = {v8f{}, v8f{}}, axn[2] = {v8f{}, v8f{}};
    v8f ahr[2] = {v8f{}, v8f{}}, ahz[2] = {v8f{}, v8f{}}, ahn[2] = {v8f{}, v8f{}};

    // x @ W_ih^T  (K = 512)
    const unsigned short* wr = wih_bf + (size_t)(jcol)       * MSG_DIM_ + koff;
    const unsigned short* wz = wih_bf + (size_t)(256 + jcol) * MSG_DIM_ + koff;
    const unsigned short* wn = wih_bf + (size_t)(512 + jcol) * MSG_DIM_ + koff;
    for (int kb = 0; kb < MSG_DIM_; kb += 32) {
      v16bf br = ldfrag(wr + kb);
      v16bf bz = ldfrag(wz + kb);
      v16bf bn = ldfrag(wn + kb);
      #pragma unroll
      for (int mt = 0; mt < 2; ++mt) {
        const unsigned short* ap = SX + (mg * 32 + mt * 16 + cl) * SX_STR + koff + kb;
        v16bf a = ldfrag(ap);
        axr[mt] = __builtin_amdgcn_wmma_f32_16x16x32_bf16(false, a, false, br, (short)0, axr[mt], false, false);
        axz[mt] = __builtin_amdgcn_wmma_f32_16x16x32_bf16(false, a, false, bz, (short)0, axz[mt], false, false);
        axn[mt] = __builtin_amdgcn_wmma_f32_16x16x32_bf16(false, a, false, bn, (short)0, axn[mt], false, false);
      }
    }
    // h @ W_hh^T  (K = 256)
    const unsigned short* vr = whh_bf + (size_t)(jcol)       * MEM_DIM_ + koff;
    const unsigned short* vz = whh_bf + (size_t)(256 + jcol) * MEM_DIM_ + koff;
    const unsigned short* vn = whh_bf + (size_t)(512 + jcol) * MEM_DIM_ + koff;
    for (int kb = 0; kb < MEM_DIM_; kb += 32) {
      v16bf br = ldfrag(vr + kb);
      v16bf bz = ldfrag(vz + kb);
      v16bf bn = ldfrag(vn + kb);
      #pragma unroll
      for (int mt = 0; mt < 2; ++mt) {
        const unsigned short* ap = SH + (mg * 32 + mt * 16 + cl) * SH_STR + koff + kb;
        v16bf a = ldfrag(ap);
        ahr[mt] = __builtin_amdgcn_wmma_f32_16x16x32_bf16(false, a, false, br, (short)0, ahr[mt], false, false);
        ahz[mt] = __builtin_amdgcn_wmma_f32_16x16x32_bf16(false, a, false, bz, (short)0, ahz[mt], false, false);
        ahn[mt] = __builtin_amdgcn_wmma_f32_16x16x32_bf16(false, a, false, bn, (short)0, ahn[mt], false, false);
      }
    }
    // fused GRU gates (output column == jcol is constant per lane);
    // write pre-LayerNorm result straight to d_out (scattered rows)
    const float bihr = b_ih[jcol], bihz = b_ih[256 + jcol], bihn = b_ih[512 + jcol];
    const float bhhr = b_hh[jcol], bhhz = b_hh[256 + jcol], bhhn = b_hh[512 + jcol];
    #pragma unroll
    for (int mt = 0; mt < 2; ++mt) {
      #pragma unroll
      for (int v = 0; v < 8; ++v) {
        int   mloc = mg * 32 + mt * 16 + v + (hi << 3);
        float rr = sigmoidf_(axr[mt][v] + bihr + ahr[mt][v] + bhhr);
        float zz = sigmoidf_(axz[mt][v] + bihz + ahz[mt][v] + bhhz);
        float nn = tanhf(axn[mt][v] + bihn + rr * (ahn[mt][v] + bhhn));
        float hv = memory_table[(size_t)ridx[mloc] * MEM_DIM_ + jcol]; // f32 h (L2 hit)
        out_mem[(size_t)orow[mloc] * MEM_DIM_ + jcol] =
            (1.0f - zz) * nn + zz * hv;
      }
    }
  }
}

// ------------------------------------------------- LayerNorm (in place)
// One wave per updated row of d_out; 8 elements per lane.
__global__ __launch_bounds__(256)
void tgn_ln_kernel(const int* __restrict__ tu_ids,
                   const float* __restrict__ gamma,
                   const float* __restrict__ beta,
                   float* __restrict__ out_mem) {
  const int lane = threadIdx.x & 31;
  const int warp = threadIdx.x >> 5;
  const int m    = blockIdx.x * 8 + warp;        // 0 .. M_UPDATE_-1
  const int row  = tu_ids[m];
  float* p = out_mem + (size_t)row * MEM_DIM_ + lane * 8;

  float4 a = *(const float4*)(p);
  float4 b = *(const float4*)(p + 4);
  float s  = a.x + a.y + a.z + a.w + b.x + b.y + b.z + b.w;
  float s2 = a.x*a.x + a.y*a.y + a.z*a.z + a.w*a.w
           + b.x*b.x + b.y*b.y + b.z*b.z + b.w*b.w;
  #pragma unroll
  for (int off = 1; off < 32; off <<= 1) {
    s  += __shfl_xor(s,  off);
    s2 += __shfl_xor(s2, off);
  }
  const float mean = s * (1.0f / 256.0f);
  const float var  = s2 * (1.0f / 256.0f) - mean * mean;
  const float rstd = rsqrtf(var + 1e-5f);

  float4 g0 = *(const float4*)(gamma + lane * 8);
  float4 g1 = *(const float4*)(gamma + lane * 8 + 4);
  float4 be0 = *(const float4*)(beta + lane * 8);
  float4 be1 = *(const float4*)(beta + lane * 8 + 4);
  float4 o0, o1;
  o0.x = (a.x - mean) * rstd * g0.x + be0.x;
  o0.y = (a.y - mean) * rstd * g0.y + be0.y;
  o0.z = (a.z - mean) * rstd * g0.z + be0.z;
  o0.w = (a.w - mean) * rstd * g0.w + be0.w;
  o1.x = (b.x - mean) * rstd * g1.x + be1.x;
  o1.y = (b.y - mean) * rstd * g1.y + be1.y;
  o1.z = (b.z - mean) * rstd * g1.z + be1.z;
  o1.w = (b.w - mean) * rstd * g1.w + be1.w;
  *(float4*)(p)     = o0;
  *(float4*)(p + 4) = o1;
}

// ----------------------------------------------------------------- launch
extern "C" void kernel_launch(void* const* d_in, const int* in_sizes, int n_in,
                              void* d_out, int out_size, void* d_ws, size_t ws_size,
                              hipStream_t stream) {
  const float* memory_table = (const float*)d_in[0];
  const float* last_update  = (const float*)d_in[1];
  const float* W_ih         = (const float*)d_in[2];
  const float* W_hh         = (const float*)d_in[3];
  const float* b_ih         = (const float*)d_in[4];
  const float* b_hh         = (const float*)d_in[5];
  const float* gamma        = (const float*)d_in[6];
  const float* beta         = (const float*)d_in[7];
  const float* messages     = (const float*)d_in[8];
  const float* timestamps   = (const float*)d_in[9];
  const int*   node_ids     = (const int*)d_in[10];
  const int*   tu_ids       = (const int*)d_in[11];
  float* out = (float*)d_out;

  unsigned short* wih_bf = (unsigned short*)d_ws;          // 768*512 bf16
  unsigned short* whh_bf = wih_bf + 768 * 512;             // 768*256 bf16

  // 1) weights f32 -> bf16 (resident in L2 afterwards)
  tgn_convert_w_kernel<<<(768 * 512 + 255) / 256, 256, 0, stream>>>(
      W_ih, W_hh, wih_bf, whh_bf);

  // 2) gather-only rows + last_update outputs
  {
    const int ROWS  = N_GATHER_ - M_UPDATE_;
    const int ELTS4 = ROWS * (MEM_DIM_ / 4);
    int total = ELTS4 > N_GATHER_ ? ELTS4 : N_GATHER_;
    tgn_tail_kernel<<<(total + 255) / 256, 256, 0, stream>>>(
        memory_table, last_update, timestamps, node_ids, out);
  }

  // 3) fused GRU via bf16 WMMA (writes pre-LN rows)
  tgn_gru_wmma_kernel<<<M_UPDATE_ / BM, TPB, SMEM_BYTES, stream>>>(
      memory_table, messages, wih_bf, whh_bf, b_ih, b_hh,
      node_ids, tu_ids, out);

  // 4) LayerNorm in place on the updated rows
  tgn_ln_kernel<<<M_UPDATE_ / 8, 256, 0, stream>>>(tu_ids, gamma, beta, out);
}
